// RoutedAllFC_61349312856156
// MI455X (gfx1250) — matrix operand order repaired and checked by reading, not compiled, not run on hardware
//
#include <hip/hip_runtime.h>
#include <cstdint>

typedef _Float16 h16;
typedef __attribute__((ext_vector_type(16))) _Float16 v16h;
typedef __attribute__((ext_vector_type(8)))  float    v8f;
typedef __attribute__((ext_vector_type(4)))  int      v4i;

union Frag {
    uint4     u[2];
    v16h      v;
    _Float16  h[16];
};

#define GEMM_WAVES 4
#define CHN 160

// ---- async direct-to-LDS (CDNA5) detection --------------------------------
#if defined(__has_builtin)
#if __has_builtin(__builtin_amdgcn_global_load_async_to_lds_b128)
#define HAVE_ASYNC_LDS 1
#endif
#if __has_builtin(__builtin_amdgcn_s_wait_asynccnt)
#define HAVE_WAIT_ASYNC 1
#endif
#endif
#ifndef HAVE_ASYNC_LDS
#define HAVE_ASYNC_LDS 0
#endif
#ifndef HAVE_WAIT_ASYNC
#define HAVE_WAIT_ASYNC 0
#endif

#if HAVE_ASYNC_LDS
// builtin signature (from hipcc diagnostic): (int4 AS1*, int4 AS3*, imm, imm)
typedef __attribute__((address_space(1))) v4i as1_v4i;
typedef __attribute__((address_space(3))) v4i as3_v4i;
#endif

__device__ __forceinline__ void wait_async5() {
#if HAVE_ASYNC_LDS
#if HAVE_WAIT_ASYNC
    __builtin_amdgcn_s_wait_asynccnt(5);
#else
    asm volatile("s_wait_asynccnt 5" ::: "memory");
#endif
#endif
}
__device__ __forceinline__ void wait_async0() {
#if HAVE_ASYNC_LDS
#if HAVE_WAIT_ASYNC
    __builtin_amdgcn_s_wait_asynccnt(0);
#else
    asm volatile("s_wait_asynccnt 0" ::: "memory");
#endif
#endif
}

// ---------------------------------------------------------------------------
// conv1 implicit GEMM (CIN=3, KPAD=32): element-wise im2col gather, 10 WMMAs.
// ---------------------------------------------------------------------------
template <int CIN, int KPAD>
__global__ void __launch_bounds__(128)
k_conv_gemm(const h16* __restrict__ act, const h16* __restrict__ wb,
            const float* __restrict__ bias, h16* __restrict__ pre,
            int H, int W, int poff)
{
    const int lane = threadIdx.x & 31;
    const int wave = threadIdx.x >> 5;
    const int r    = lane & 15;
    const int half = lane >> 4;

    const int tileBase = (blockIdx.x * GEMM_WAVES + wave) * 16;
    const int mg = poff + tileBase + r;
    const int HW = H * W;
    const int b  = mg / HW;
    const int rm = mg - b * HW;
    const int y  = rm / W;
    const int x  = rm - y * W;

    v8f acc[10];
#pragma unroll
    for (int i = 0; i < 10; i++) {
        v8f z = {0.f, 0.f, 0.f, 0.f, 0.f, 0.f, 0.f, 0.f};
        acc[i] = z;
    }

    for (int kk = 0; kk < KPAD; kk += 32) {
        Frag a;
#pragma unroll
        for (int j = 0; j < 16; j++) {
            const int k = kk + half * 16 + j;
            _Float16 val = (_Float16)0.f;
            if (k < 9 * CIN) {
                const int s  = k / CIN, c = k - s * CIN;
                const int yy = y + s / 3 - 1, xx = x + s % 3 - 1;
                if (yy >= 0 && yy < H && xx >= 0 && xx < W)
                    val = act[((size_t)(b * H + yy) * W + xx) * CIN + c];
            }
            a.h[j] = val;
        }
#pragma unroll
        for (int nt = 0; nt < 10; nt++) {
            const h16* bp = wb + (size_t)(nt * 16 + r) * KPAD + kk + half * 16;
            Frag bf;
            bf.u[0] = ((const uint4*)bp)[0];
            bf.u[1] = ((const uint4*)bp)[1];
            acc[nt] = __builtin_amdgcn_wmma_f32_16x16x32_f16(
                false, a.v, false, bf.v, (short)0, acc[nt], false, false);
        }
    }

    const int mloc = tileBase + half * 8;
#pragma unroll
    for (int nt = 0; nt < 10; nt++) {
        const int n  = nt * 16 + r;
        const float bn = bias[n];
#pragma unroll
        for (int g = 0; g < 8; g++) {
            float v = acc[nt][g] + bn;
            v = v > 0.f ? v : 0.f;
            pre[(size_t)(mloc + g) * CHN + n] = (h16)v;
        }
    }
}

// ---------------------------------------------------------------------------
// conv2-4 implicit GEMM (CIN=160, K=1440), LDS-staged weights.
//  block = 128 thr (4 waves) covers 64 M rows x 160 N.
//  wave: M-subtile (wave>>1)*32 rows, N-half (wave&1)*80.
//  Per K-step(32): B slice 160x32 f16 (10 KB) double-buffered in LDS via
//  async direct-to-LDS (ASYNCcnt) when available; fragments read as
//  ds_load_b128. A = bounds-checked global b128 (NHWC, tap-major K order).
// ---------------------------------------------------------------------------
__global__ void __launch_bounds__(128)
k_conv160(const h16* __restrict__ act, const h16* __restrict__ wb,
          const float* __restrict__ bias, h16* __restrict__ pre,
          int H, int W, int poff)
{
    __shared__ h16 sb[2][5120];                 // 2 x 10 KB: [n][32 k] per buffer

    const int tid  = threadIdx.x;
    const int lane = tid & 31;
    const int wave = tid >> 5;
    const int r    = lane & 15;
    const int half = lane >> 4;
    const int mtile = wave >> 1;                // 0..1
    const int n0    = (wave & 1) * 80;          // N half

    const int mbase = poff + blockIdx.x * 64 + mtile * 32;
    const int HW = H * W;

    int by[2], bx[2], bi[2];                    // per-lane coords of the 2 A rows
#pragma unroll
    for (int mi = 0; mi < 2; mi++) {
        const int mg = mbase + mi * 16 + r;
        const int b  = mg / HW;
        const int rm = mg - b * HW;
        by[mi] = rm / W;
        bx[mi] = rm - by[mi] * W;
        bi[mi] = b;
    }

    v8f acc[2][5];
#pragma unroll
    for (int mi = 0; mi < 2; mi++)
#pragma unroll
        for (int nt = 0; nt < 5; nt++) {
            v8f z = {0.f, 0.f, 0.f, 0.f, 0.f, 0.f, 0.f, 0.f};
            acc[mi][nt] = z;
        }
    const uint4 zero4 = make_uint4(0u, 0u, 0u, 0u);

    // stage B k-slice [160 x 32] into LDS buffer: 640 16B chunks, 5 per thread
    auto stage = [&](int buf, int kkk) {
#pragma unroll
        for (int i = 0; i < 5; i++) {
            const int c    = tid + i * 128;
            const int n    = c >> 2;
            const int part = c & 3;
            const h16* g = wb + (size_t)n * 1440 + kkk + part * 8;
            h16* l = &sb[buf][n * 32 + part * 8];
#if HAVE_ASYNC_LDS
            __builtin_amdgcn_global_load_async_to_lds_b128(
                (as1_v4i*)(uintptr_t)g, (as3_v4i*)(uint32_t)(uintptr_t)l, 0, 0);
#else
            *(uint4*)l = *(const uint4*)g;
#endif
        }
    };

    stage(0, 0);
    int buf = 0;
    for (int kk = 0; kk < 1440; kk += 32) {
        const int nxt = kk + 32;
        if (nxt < 1440) { stage(buf ^ 1, nxt); wait_async5(); }
        else            { wait_async0(); }
        __syncthreads();

        // A fragments: taps shared across both M rows; runs of 8 stay inside a tap
        Frag a[2];
        {
            const int k1 = kk + half * 8;
            const int s1 = k1 / CHN, c1 = k1 - s1 * CHN;
            const int dy1 = s1 / 3 - 1, dx1 = s1 % 3 - 1;
            const int k2 = k1 + 16;
            const int s2 = k2 / CHN, c2 = k2 - s2 * CHN;
            const int dy2 = s2 / 3 - 1, dx2 = s2 % 3 - 1;
#pragma unroll
            for (int mi = 0; mi < 2; mi++) {
                const int y1 = by[mi] + dy1, x1 = bx[mi] + dx1;
                a[mi].u[0] = (y1 >= 0 && y1 < H && x1 >= 0 && x1 < W)
                    ? *(const uint4*)(act + ((size_t)(bi[mi] * H + y1) * W + x1) * CHN + c1)
                    : zero4;
                const int y2 = by[mi] + dy2, x2 = bx[mi] + dx2;
                a[mi].u[1] = (y2 >= 0 && y2 < H && x2 >= 0 && x2 < W)
                    ? *(const uint4*)(act + ((size_t)(bi[mi] * H + y2) * W + x2) * CHN + c2)
                    : zero4;
            }
        }

#pragma unroll
        for (int nt = 0; nt < 5; nt++) {
            const h16* lp = &sb[buf][(n0 + nt * 16 + r) * 32 + half * 8];
            Frag bf;
            bf.u[0] = ((const uint4*)lp)[0];
            bf.u[1] = ((const uint4*)lp)[2];
            acc[0][nt] = __builtin_amdgcn_wmma_f32_16x16x32_f16(
                false, a[0].v, false, bf.v, (short)0, acc[0][nt], false, false);
            acc[1][nt] = __builtin_amdgcn_wmma_f32_16x16x32_f16(
                false, a[1].v, false, bf.v, (short)0, acc[1][nt], false, false);
        }
        __syncthreads();                       // protect buf before overwrite
        buf ^= 1;
    }

    // epilogue: bias + ReLU -> chunk-local pre-pool [m, 160] f16
    const int mloc0 = blockIdx.x * 64 + mtile * 32 + half * 8;
#pragma unroll
    for (int nt = 0; nt < 5; nt++) {
        const int n  = n0 + nt * 16 + r;
        const float bn = bias[n];
#pragma unroll
        for (int mi = 0; mi < 2; mi++) {
            const int mrow = mloc0 + mi * 16;
#pragma unroll
            for (int g = 0; g < 8; g++) {
                float v = acc[mi][nt][g] + bn;
                v = v > 0.f ? v : 0.f;
                pre[(size_t)(mrow + g) * CHN + n] = (h16)v;
            }
        }
    }
}

// ---------------------------------------------------------------------------
// Dense GEMM for the expert FCs:  C[z][m, n] = A[m, :] . B[z][n, :]  (+bias)
// ---------------------------------------------------------------------------
template <int NT, bool RELU, bool F32OUT>
__global__ void __launch_bounds__(128)
k_fc_gemm(const h16* __restrict__ A, const h16* __restrict__ Bw,
          const float* __restrict__ bias, void* __restrict__ Cv,
          int M, int Ntot, int K,
          size_t strideBz, int strideBiasZ, size_t strideCz)
{
    const int lane = threadIdx.x & 31;
    const int wave = threadIdx.x >> 5;
    const int r    = lane & 15;
    const int half = lane >> 4;
    const int m0   = (blockIdx.x * GEMM_WAVES + wave) * 16;
    if (m0 >= M) return;                                   // wave-uniform
    const int n0 = blockIdx.y * (NT * 16);
    const int z  = blockIdx.z;

    const h16* Ar = A + (size_t)(m0 + r) * K;
    const h16* Bz = Bw + (size_t)z * strideBz;

    v8f acc[NT];
#pragma unroll
    for (int i = 0; i < NT; i++) {
        v8f zz = {0.f, 0.f, 0.f, 0.f, 0.f, 0.f, 0.f, 0.f};
        acc[i] = zz;
    }

    for (int kk = 0; kk < K; kk += 32) {
        Frag a;
        a.u[0] = *(const uint4*)(Ar + kk + half * 8);
        a.u[1] = *(const uint4*)(Ar + kk + 16 + half * 8);
#pragma unroll
        for (int nt = 0; nt < NT; nt++) {
            const h16* bp = Bz + (size_t)(n0 + nt * 16 + r) * K + kk + half * 8;
            Frag bf;
            bf.u[0] = ((const uint4*)bp)[0];
            bf.u[1] = ((const uint4*)bp)[2];
            acc[nt] = __builtin_amdgcn_wmma_f32_16x16x32_f16(
                false, a.v, false, bf.v, (short)0, acc[nt], false, false);
        }
    }

    const float* biasz = bias + (size_t)z * strideBiasZ;
#pragma unroll
    for (int nt = 0; nt < NT; nt++) {
        const int n = n0 + nt * 16 + r;
        const float bn = biasz[n];
#pragma unroll
        for (int g = 0; g < 8; g++) {
            const int m = m0 + half * 8 + g;
            float v = acc[nt][g] + bn;
            if (RELU) v = v > 0.f ? v : 0.f;
            if (F32OUT)
                ((float*)Cv)[(size_t)z * strideCz + (size_t)m * Ntot + n] = v;
            else
                ((h16*)Cv)[(size_t)z * strideCz + (size_t)m * Ntot + n] = (h16)v;
        }
    }
}

// ---------------------------------------------------------------------------
// 2x2 maxpool over chunk-local pre-pool [pcnt, 160] -> global NHWC pooled out
// ---------------------------------------------------------------------------
__global__ void k_pool(const h16* __restrict__ pre, h16* __restrict__ out,
                       int H, int W, int poff, int pcnt)
{
    const int tid   = blockIdx.x * blockDim.x + threadIdx.x;
    const int total = (pcnt / 4) * CHN;
    if (tid >= total) return;
    const int c  = tid % CHN;
    const int pp = tid / CHN;
    const int q  = poff / 4 + pp;
    const int Hp = H / 2, Wp = W / 2;
    const int b  = q / (Hp * Wp);
    const int rq = q - b * (Hp * Wp);
    const int ph = rq / Wp, pw = rq - ph * Wp;
    const size_t mg = (size_t)(b * H + ph * 2) * W + pw * 2;
    const size_t l  = mg - poff;
    const float v00 = (float)pre[(l)         * CHN + c];
    const float v01 = (float)pre[(l + 1)     * CHN + c];
    const float v10 = (float)pre[(l + W)     * CHN + c];
    const float v11 = (float)pre[(l + W + 1) * CHN + c];
    const float m = fmaxf(fmaxf(v00, v01), fmaxf(v10, v11));
    out[((size_t)(b * Hp + ph) * Wp + pw) * CHN + c] = (h16)m;
}

// ---------------------------------------------------------------------------
// Layout / conversion kernels
// ---------------------------------------------------------------------------
__global__ void k_x_to_nhwc(const float* __restrict__ x, h16* __restrict__ xh)
{
    const int tid = blockIdx.x * blockDim.x + threadIdx.x;
    if (tid >= 64 * 64 * 64 * 3) return;
    const int c = tid % 3;
    int t = tid / 3;
    const int w = t % 64; t /= 64;
    const int h = t % 64;
    const int b = t / 64;
    xh[tid] = (h16)x[(((size_t)b * 3 + c) * 64 + h) * 64 + w];
}

__global__ void k_repack_w(const float* __restrict__ Wsrc, h16* __restrict__ wb,
                           int Cin, int Kpad)
{
    const int tid = blockIdx.x * blockDim.x + threadIdx.x;
    if (tid >= CHN * Kpad) return;
    const int n = tid / Kpad, k = tid - n * Kpad;
    float v = 0.f;
    if (k < 9 * Cin) {
        const int s = k / Cin, c = k - s * Cin;
        const int dr = s / 3, dc = s % 3;
        v = Wsrc[(((size_t)n * Cin + c) * 3 + dr) * 3 + dc];
    }
    wb[tid] = (h16)v;
}

__global__ void k_f32_to_f16(const float* __restrict__ in, h16* __restrict__ out, int n)
{
    const int tid = blockIdx.x * blockDim.x + threadIdx.x;
    if (tid < n) out[tid] = (h16)in[tid];
}

__global__ void k_pad_e3(const float* __restrict__ W, const float* __restrict__ bsrc,
                         h16* __restrict__ Wp, float* __restrict__ bp)
{
    const int tid = blockIdx.x * blockDim.x + threadIdx.x;
    if (tid >= 8 * 16 * 320) return;
    const int e  = tid / (16 * 320);
    const int rr = tid - e * (16 * 320);
    const int n  = rr / 320, k = rr - n * 320;
    Wp[tid] = (h16)((n < 10) ? W[((size_t)e * 10 + n) * 320 + k] : 0.f);
    if (k == 0) bp[e * 16 + n] = (n < 10) ? bsrc[e * 10 + n] : 0.f;
}

__global__ void k_bn_stats(const h16* __restrict__ act, const float* __restrict__ gam,
                           const float* __restrict__ bet, float* __restrict__ coef)
{
    const int c = blockIdx.x * blockDim.x + threadIdx.x;
    if (c >= CHN) return;
    float s = 0.f, s2 = 0.f;
    for (int i = 0; i < 1024; i++) {
        const float v = (float)act[(size_t)i * CHN + c];
        s += v; s2 += v * v;
    }
    const float mean = s * (1.f / 1024.f);
    const float var  = s2 * (1.f / 1024.f) - mean * mean;
    const float rs   = rsqrtf(var + 1e-5f);
    const float a    = rs * gam[c];
    coef[c]       = a;
    coef[CHN + c] = bet[c] - mean * a;
}

__global__ void k_bn_apply(const h16* __restrict__ act, const float* __restrict__ coef,
                           h16* __restrict__ fcin)
{
    const int tid = blockIdx.x * blockDim.x + threadIdx.x;
    if (tid >= 64 * 2560) return;
    const int b = tid / 2560, f = tid - b * 2560;
    const int c = f / 16, p = f - c * 16;
    const float v = (float)act[((size_t)b * 16 + p) * CHN + c];
    fcin[tid] = (h16)(v * coef[c] + coef[CHN + c]);
}

__global__ void k_gather16(const h16* __restrict__ G, const int* __restrict__ a,
                           h16* __restrict__ Z, int D)
{
    const int tid = blockIdx.x * blockDim.x + threadIdx.x;
    if (tid >= 64 * D) return;
    const int b = tid / D, j = tid - b * D;
    Z[tid] = G[((size_t)a[b] * 64 + b) * D + j];
}

__global__ void k_gather_out(const float* __restrict__ G3, const int* __restrict__ a,
                             float* __restrict__ out)
{
    const int tid = blockIdx.x * blockDim.x + threadIdx.x;
    if (tid >= 640) return;
    const int b = tid / 10, o = tid - b * 10;
    out[tid] = G3[((size_t)a[b] * 64 + b) * 16 + o];
}

// ---------------------------------------------------------------------------
static inline int cdiv(int a, int b) { return (a + b - 1) / b; }

extern "C" void kernel_launch(void* const* d_in, const int* in_sizes, int n_in,
                              void* d_out, int out_size, void* d_ws, size_t ws_size,
                              hipStream_t stream)
{
    (void)in_sizes; (void)n_in; (void)out_size; (void)ws_size;
    const float* x    = (const float*)d_in[0];
    const int*   a1   = (const int*)d_in[1];
    const int*   a2   = (const int*)d_in[2];
    const int*   a3   = (const int*)d_in[3];
    const float* W1   = (const float*)d_in[4];  const float* b1 = (const float*)d_in[5];
    const float* W2   = (const float*)d_in[6];  const float* b2 = (const float*)d_in[7];
    const float* W3   = (const float*)d_in[8];  const float* b3 = (const float*)d_in[9];
    const float* W4   = (const float*)d_in[10]; const float* b4 = (const float*)d_in[11];
    const float* gam  = (const float*)d_in[12];
    const float* bet  = (const float*)d_in[13];
    const float* E1W  = (const float*)d_in[14]; const float* E1b = (const float*)d_in[15];
    const float* E2W  = (const float*)d_in[16]; const float* E2b = (const float*)d_in[17];
    const float* E3W  = (const float*)d_in[18]; const float* E3b = (const float*)d_in[19];

    char* p = (char*)d_ws;
    auto alloc = [&](size_t bytes) -> char* {
        char* q = p;
        p += (bytes + 255) & ~(size_t)255;
        return q;
    };
    h16*   XH    = (h16*)  alloc((size_t)786432 * 2);
    h16*   WB1   = (h16*)  alloc((size_t)160 * 32 * 2);
    h16*   WB2   = (h16*)  alloc((size_t)160 * 1440 * 2);
    h16*   WB3   = (h16*)  alloc((size_t)160 * 1440 * 2);
    h16*   WB4   = (h16*)  alloc((size_t)160 * 1440 * 2);
    h16*   E1Wh  = (h16*)  alloc((size_t)6553600 * 2);
    h16*   E2Wh  = (h16*)  alloc((size_t)819200 * 2);
    h16*   E3Wp  = (h16*)  alloc((size_t)40960 * 2);
    float* E3bp  = (float*)alloc((size_t)128 * 4);
    h16*   PRE   = (h16*)  alloc((size_t)16384 * 160 * 2);
    h16*   ACTA  = (h16*)  alloc((size_t)10485760 * 2);
    h16*   ACTB  = (h16*)  alloc((size_t)2621440 * 2);
    float* BNC   = (float*)alloc((size_t)320 * 4);
    h16*   FCIN  = (h16*)  alloc((size_t)163840 * 2);
    h16*   G1    = (h16*)  alloc((size_t)163840 * 2);
    h16*   Z1    = (h16*)  alloc((size_t)20480 * 2);
    h16*   G2    = (h16*)  alloc((size_t)163840 * 2);
    h16*   Z2    = (h16*)  alloc((size_t)20480 * 2);
    float* G3    = (float*)alloc((size_t)8192 * 4);

    k_x_to_nhwc<<<cdiv(786432, 256), 256, 0, stream>>>(x, XH);
    k_repack_w<<<cdiv(160 * 32, 256),   256, 0, stream>>>(W1, WB1, 3,   32);
    k_repack_w<<<cdiv(160 * 1440, 256), 256, 0, stream>>>(W2, WB2, 160, 1440);
    k_repack_w<<<cdiv(160 * 1440, 256), 256, 0, stream>>>(W3, WB3, 160, 1440);
    k_repack_w<<<cdiv(160 * 1440, 256), 256, 0, stream>>>(W4, WB4, 160, 1440);
    k_f32_to_f16<<<cdiv(6553600, 256), 256, 0, stream>>>(E1W, E1Wh, 6553600);
    k_f32_to_f16<<<cdiv(819200, 256),  256, 0, stream>>>(E2W, E2Wh, 819200);
    k_pad_e3<<<cdiv(8 * 16 * 320, 256), 256, 0, stream>>>(E3W, E3b, E3Wp, E3bp);

    // conv1: 64x64, M = 262144 (chunks of 16384 pixels = 4 images)
    for (int off = 0; off < 262144; off += 16384) {
        k_conv_gemm<3, 32><<<dim3(16384 / 64), 128, 0, stream>>>(XH, WB1, b1, PRE, 64, 64, off);
        k_pool<<<cdiv(4096 * CHN, 256), 256, 0, stream>>>(PRE, ACTA, 64, 64, off, 16384);
    }
    // conv2: 32x32, M = 65536
    for (int off = 0; off < 65536; off += 16384) {
        k_conv160<<<dim3(16384 / 64), 128, 0, stream>>>(ACTA, WB2, b2, PRE, 32, 32, off);
        k_pool<<<cdiv(4096 * CHN, 256), 256, 0, stream>>>(PRE, ACTB, 32, 32, off, 16384);
    }
    // conv3: 16x16, M = 16384
    k_conv160<<<dim3(16384 / 64), 128, 0, stream>>>(ACTB, WB3, b3, PRE, 16, 16, 0);
    k_pool<<<cdiv(4096 * CHN, 256), 256, 0, stream>>>(PRE, ACTA, 16, 16, 0, 16384);
    // conv4: 8x8, M = 4096
    k_conv160<<<dim3(4096 / 64), 128, 0, stream>>>(ACTA, WB4, b4, PRE, 8, 8, 0);
    k_pool<<<cdiv(1024 * CHN, 256), 256, 0, stream>>>(PRE, ACTB, 8, 8, 0, 4096);

    k_bn_stats<<<1, 160, 0, stream>>>(ACTB, gam, bet, BNC);
    k_bn_apply<<<cdiv(64 * 2560, 256), 256, 0, stream>>>(ACTB, BNC, FCIN);

    k_fc_gemm<10, true, false><<<dim3(1, 2, 8), 128, 0, stream>>>(
        FCIN, E1Wh, E1b, G1, 64, 320, 2560, (size_t)819200, 320, (size_t)20480);
    k_gather16<<<cdiv(64 * 320, 256), 256, 0, stream>>>(G1, a1, Z1, 320);
    k_fc_gemm<10, true, false><<<dim3(1, 2, 8), 128, 0, stream>>>(
        Z1, E2Wh, E2b, G2, 64, 320, 320, (size_t)102400, 320, (size_t)20480);
    k_gather16<<<cdiv(64 * 320, 256), 256, 0, stream>>>(G2, a2, Z2, 320);
    k_fc_gemm<1, false, true><<<dim3(1, 1, 8), 128, 0, stream>>>(
        Z2, E3Wp, E3bp, G3, 64, 16, 320, (size_t)5120, 16, (size_t)1024);
    k_gather_out<<<cdiv(640, 256), 256, 0, stream>>>(G3, a3, (float*)d_out);
}
